// PiCANet_G_39968965656961
// MI455X (gfx1250) — matrix operand order repaired and checked by reading, not compiled, not run on hardware
//
#include <hip/hip_runtime.h>

// Problem constants (B=64, C=512, S=28, HID=256)
#define SDIM 28
#define NB   1792          // B*S : recurrent batch for both LSTM stages
#define KIN  512           // input feature size for both stages (C and 2*HID)
#define HIDN 256
#define G4   1024          // 4*HID
#define NPIX 50176         // 64*28*28

// Padded LDS row strides (elements): +8 bf16 = +16B so that row r of a
// fragment read starts at bank 4r (mod 64) -> conflict-free ds_load_b128.
#define XLDS_STRIDE 520    // 512 + 8
#define HLDS_STRIDE 264    // 256 + 8

typedef __attribute__((ext_vector_type(16))) __bf16 v16bf;
typedef __attribute__((ext_vector_type(8)))  __bf16 v8bf;
typedef __attribute__((ext_vector_type(8)))  float  v8f;

static __device__ __forceinline__ __bf16 f2bf(float f) {
  unsigned u = __builtin_bit_cast(unsigned, f);
  unsigned r = (u + 0x7FFFu + ((u >> 16) & 1u)) >> 16;   // round-to-nearest-even
  unsigned short s = (unsigned short)r;
  return __builtin_bit_cast(__bf16, s);
}
static __device__ __forceinline__ float bf2f(__bf16 h) {
  unsigned short s = __builtin_bit_cast(unsigned short, h);
  unsigned u = ((unsigned)s) << 16;
  return __builtin_bit_cast(float, u);
}
static __device__ __forceinline__ float sigmoidf_(float x) {
  return 1.0f / (1.0f + __expf(-x));
}
static __device__ __forceinline__ v8f wmma_bf16(v16bf a, v16bf b, v8f c) {
  // D(f32 16x16) = A(bf16 16x32) * B(bf16 32x16) + C
  return __builtin_amdgcn_wmma_f32_16x16x32_bf16(false, a, false, b, (short)0, c,
                                                 false, false);
}
// A fragment: lane L (row = L&15) holds K = [16*half, 16*half+16)
static __device__ __forceinline__ v16bf load_a_frag(const __bf16* row, int k0, int half) {
  union { v16bf v; v8bf h[2]; } u;
  u.h[0] = *(const v8bf*)(row + k0 + half * 16);
  u.h[1] = *(const v8bf*)(row + k0 + half * 16 + 8);
  return u.v;
}
// Async global -> LDS copy of one 16-byte chunk (GLOBAL_LOAD_ASYNC_TO_LDS_B128,
// GV addressing; tracked by ASYNCcnt).
static __device__ __forceinline__ void async_load_b128(unsigned lds_off,
                                                       const __bf16* gptr) {
  asm volatile("global_load_async_to_lds_b128 %0, %1, off"
               : : "v"(lds_off), "v"(gptr) : "memory");
}
static __device__ __forceinline__ void wait_asynccnt0() {
  asm volatile("s_wait_asynccnt 0x0" ::: "memory");
}

// ---------------------------------------------------------------------------
// Utility kernels
// ---------------------------------------------------------------------------
__global__ void cvt_bf16_kernel(const float* __restrict__ s, __bf16* __restrict__ d, int n) {
  int i = blockIdx.x * 256 + threadIdx.x;
  if (i < n) d[i] = f2bf(s[i]);
}

// Xv[t=w][n=b*28+h][c] = x[b][c][h][w]  (vertical scan layout, bf16)
__global__ void pack_xv_kernel(const float* __restrict__ x, __bf16* __restrict__ Xv) {
  size_t i = (size_t)blockIdx.x * 256 + threadIdx.x;
  const size_t total = (size_t)SDIM * NB * KIN;
  if (i >= total) return;
  int c = (int)(i & (KIN - 1));
  size_t r = i >> 9;                 // t*NB + n
  int n = (int)(r % NB);
  int t = (int)(r / NB);             // w
  int b = n / SDIM, h = n % SDIM;
  Xv[i] = f2bf(x[(((size_t)b * KIN + c) * SDIM + h) * SDIM + t]);
}

// ---------------------------------------------------------------------------
// Persistent bidirectional-LSTM stage kernel: one workgroup (16 waves) owns
// all hidden columns of one (row-tile, direction) pair, so the 28-step
// recurrence closes entirely inside the workgroup:
//   - h(t) lives in LDS (ldsH), rewritten each step behind a barrier
//   - c    lives in one persistent v8f accumulator register per wave
//   - X(t) tile double-buffered in LDS via async global->LDS loads,
//     prefetching step t+1 while step t computes
//   gates = X[t_eff] @ Wih^T + h @ Whh^T + bias  (torch gate order i,f,g,o)
//   c' = sig(f)*c + sig(i)*tanh(g); h' = sig(o)*tanh(c')
// mode 0: scatter h into Xh (vertical -> horizontal transpose layout)
// mode 1: scatter h into pixel-major feature buffer for the attention stage
// ---------------------------------------------------------------------------
__global__ __launch_bounds__(512) void lstm_stage_kernel(
    const __bf16* __restrict__ X,                                     // [28][NB][KIN]
    const __bf16* __restrict__ WihF, const __bf16* __restrict__ WihB, // [G4][KIN]
    const __bf16* __restrict__ WhhF, const __bf16* __restrict__ WhhB, // [G4][HIDN]
    const float*  __restrict__ BiasF, const float* __restrict__ BiasB,// [G4]
    __bf16* __restrict__ Xnext,
    int mode)
{
  __shared__ __bf16 ldsX[2][16 * XLDS_STRIDE];   // double-buffered input tile
  __shared__ __bf16 ldsH[16 * HLDS_STRIDE];      // recurrent hidden state

  const int lane = threadIdx.x & 31;
  const int wv   = threadIdx.x >> 5;         // 0..15 : hidden-column tile
  const int m0   = blockIdx.x * 16;          // row tile: [0, 1792)
  const int dir  = blockIdx.y;               // 0 = forward, 1 = backward
  const int j0   = wv * 16;

  const __bf16* Wih  = dir ? WihB : WihF;
  const __bf16* Whh  = dir ? WhhB : WhhF;
  const float*  Bias = dir ? BiasB : BiasF;

  const int col   = lane & 15;               // A row offset / B column / D column
  const int khalf = lane >> 4;

  // zero the recurrent hidden state
  {
    unsigned* hz = (unsigned*)ldsH;
    for (int i = threadIdx.x; i < (16 * HLDS_STRIDE) / 2; i += 512) hz[i] = 0u;
  }

  // async staging of the X tile for timestep t into buffer `buf`
  auto stage_x = [&](int t, int buf) {
    const int te = dir ? (SDIM - 1 - t) : t;
    const __bf16* Xtile = X + ((size_t)te * NB + m0) * KIN;   // 16 x 512
    const unsigned base = (unsigned)(uintptr_t)(&ldsX[buf][0]);
    #pragma unroll
    for (int it = 0; it < 2; ++it) {                 // 1024 chunks / 512 threads
      const int ci  = it * 512 + threadIdx.x;
      const int row = ci >> 6;                       // 64 chunks per row
      const int cw  = ci & 63;
      async_load_b128(base + row * (XLDS_STRIDE * 2) + cw * 16,
                      Xtile + (size_t)row * KIN + cw * 8);
    }
  };

  stage_x(0, 0);
  wait_asynccnt0();
  __syncthreads();

  // per-wave weight row pointers (lane's B column = weight row j0+col)
  const __bf16* W0 = Wih + (size_t)(0 * HIDN + j0 + col) * KIN;
  const __bf16* W1 = Wih + (size_t)(1 * HIDN + j0 + col) * KIN;
  const __bf16* W2 = Wih + (size_t)(2 * HIDN + j0 + col) * KIN;
  const __bf16* W3 = Wih + (size_t)(3 * HIDN + j0 + col) * KIN;
  const __bf16* U0 = Whh + (size_t)(0 * HIDN + j0 + col) * HIDN;
  const __bf16* U1 = Whh + (size_t)(1 * HIDN + j0 + col) * HIDN;
  const __bf16* U2 = Whh + (size_t)(2 * HIDN + j0 + col) * HIDN;
  const __bf16* U3 = Whh + (size_t)(3 * HIDN + j0 + col) * HIDN;
  const float bi = Bias[0 * HIDN + j0 + col];
  const float bf = Bias[1 * HIDN + j0 + col];
  const float bg = Bias[2 * HIDN + j0 + col];
  const float bo = Bias[3 * HIDN + j0 + col];

  v8f creg = {};                         // persistent cell state (D layout)

  for (int t = 0; t < SDIM; ++t) {
    const int buf = t & 1;
    if (t + 1 < SDIM) stage_x(t + 1, buf ^ 1);   // overlap with compute

    const int te = dir ? (SDIM - 1 - t) : t;
    v8f acc0 = {}, acc1 = {}, acc2 = {}, acc3 = {};

    // ---- input-to-hidden GEMM (K = 512), A from LDS, reg double-buffered ----
    {
      const __bf16* Xl = &ldsX[buf][0] + (size_t)col * XLDS_STRIDE;
      int kb = khalf * 16;
      v16bf b0 = *(const v16bf*)(W0 + kb);
      v16bf b1 = *(const v16bf*)(W1 + kb);
      v16bf b2 = *(const v16bf*)(W2 + kb);
      v16bf b3 = *(const v16bf*)(W3 + kb);
      v16bf a  = load_a_frag(Xl, 0, khalf);
      #pragma unroll
      for (int k0 = 0; k0 < KIN; k0 += 32) {
        v16bf na = a, n0 = b0, n1 = b1, n2 = b2, n3 = b3;
        if (k0 + 32 < KIN) {
          const int kn = k0 + 32 + khalf * 16;
          n0 = *(const v16bf*)(W0 + kn);
          n1 = *(const v16bf*)(W1 + kn);
          n2 = *(const v16bf*)(W2 + kn);
          n3 = *(const v16bf*)(W3 + kn);
          na = load_a_frag(Xl, k0 + 32, khalf);
        }
        acc0 = wmma_bf16(a, b0, acc0);
        acc1 = wmma_bf16(a, b1, acc1);
        acc2 = wmma_bf16(a, b2, acc2);
        acc3 = wmma_bf16(a, b3, acc3);
        a = na; b0 = n0; b1 = n1; b2 = n2; b3 = n3;
      }
    }

    // ---- hidden-to-hidden GEMM (K = 256), A = h(t-1) from LDS ----
    {
      const __bf16* Hl = ldsH + (size_t)col * HLDS_STRIDE;
      int kb = khalf * 16;
      v16bf b0 = *(const v16bf*)(U0 + kb);
      v16bf b1 = *(const v16bf*)(U1 + kb);
      v16bf b2 = *(const v16bf*)(U2 + kb);
      v16bf b3 = *(const v16bf*)(U3 + kb);
      v16bf a  = load_a_frag(Hl, 0, khalf);
      #pragma unroll
      for (int k0 = 0; k0 < HIDN; k0 += 32) {
        v16bf na = a, n0 = b0, n1 = b1, n2 = b2, n3 = b3;
        if (k0 + 32 < HIDN) {
          const int kn = k0 + 32 + khalf * 16;
          n0 = *(const v16bf*)(U0 + kn);
          n1 = *(const v16bf*)(U1 + kn);
          n2 = *(const v16bf*)(U2 + kn);
          n3 = *(const v16bf*)(U3 + kn);
          na = load_a_frag(Hl, k0 + 32, khalf);
        }
        acc0 = wmma_bf16(a, b0, acc0);
        acc1 = wmma_bf16(a, b1, acc1);
        acc2 = wmma_bf16(a, b2, acc2);
        acc3 = wmma_bf16(a, b3, acc3);
        a = na; b0 = n0; b1 = n1; b2 = n2; b3 = n3;
      }
    }

    // ---- LSTM nonlinearity in the D-fragment layout ----
    // element e -> row = 8*khalf + e, col = lane&15
    __bf16 hb[8];
    #pragma unroll
    for (int e = 0; e < 8; ++e) {
      const float iv = sigmoidf_(acc0[e] + bi);
      const float fv = sigmoidf_(acc1[e] + bf);
      const float gv = tanhf   (acc2[e] + bg);
      const float ov = sigmoidf_(acc3[e] + bo);
      const float cn = fv * creg[e] + iv * gv;
      creg[e] = cn;
      hb[e] = f2bf(ov * tanhf(cn));
    }

    // scatter h into the next stage's input layout (global)
    #pragma unroll
    for (int e = 0; e < 8; ++e) {
      const int r = m0 + khalf * 8 + e;        // recurrent batch index
      const int j = j0 + col;                  // hidden unit
      const int bb = r / SDIM, loc = r % SDIM;
      size_t dst;
      if (mode == 0) {
        // vertical stage: n=(b,h), t=w  ->  Xh[h][b*28+w][dir*256+j]
        dst = (((size_t)loc * NB) + (size_t)bb * SDIM + te) * KIN + (size_t)dir * HIDN + j;
      } else {
        // horizontal stage: n=(b,w), t=h -> Hpix[(b*28+h)*28+w][dir*256+j]
        dst = (((size_t)(bb * SDIM + te)) * SDIM + loc) * KIN + (size_t)dir * HIDN + j;
      }
      Xnext[dst] = hb[e];
    }

    __syncthreads();                           // all waves done reading ldsH
    #pragma unroll
    for (int e = 0; e < 8; ++e) {              // publish h(t) for next step
      const int rl = khalf * 8 + e;
      ldsH[rl * HLDS_STRIDE + j0 + col] = hb[e];
    }
    if (t + 1 < SDIM) wait_asynccnt0();        // next X tile landed in LDS
    __syncthreads();                           // h + X visible to all waves
  }
}

// ---------------------------------------------------------------------------
// Attention head: per pixel p = (b*28+h)*28+w
//   logits[o] = feat . conv_w[o] + conv_b[o]; soft = softmax(logits)
//   out[p][c] = sum_o soft[o] * x[b][c][(o/10)*3][(o%10)*3]
// ---------------------------------------------------------------------------
__global__ __launch_bounds__(128) void attention_kernel(
    const __bf16* __restrict__ Hpix,   // [NPIX][512]
    const float* __restrict__ convw,   // [100][512]
    const float* __restrict__ convb,   // [100]
    const float* __restrict__ x,       // [64][512][28][28]
    float* __restrict__ out)           // [NPIX][512] (matches faithful reshape)
{
  const int p = blockIdx.x;
  const int b = p / (SDIM * SDIM);
  __shared__ float hv[KIN];
  __shared__ float soft[100];
  __shared__ float inv_sum;

  for (int f = threadIdx.x; f < KIN; f += 128)
    hv[f] = bf2f(Hpix[(size_t)p * KIN + f]);
  __syncthreads();

  if (threadIdx.x < 100) {
    const int o = threadIdx.x;
    const float* w = convw + (size_t)o * KIN;
    float s = convb[o];
    #pragma unroll 8
    for (int f = 0; f < KIN; ++f) s += hv[f] * w[f];
    soft[o] = s;
  }
  __syncthreads();

  if (threadIdx.x == 0) {
    float m = soft[0];
    for (int o = 1; o < 100; ++o) m = fmaxf(m, soft[o]);
    float s = 0.0f;
    for (int o = 0; o < 100; ++o) { soft[o] = __expf(soft[o] - m); s += soft[o]; }
    inv_sum = 1.0f / s;
  }
  __syncthreads();
  if (threadIdx.x < 100) soft[threadIdx.x] *= inv_sum;
  __syncthreads();

  for (int c = threadIdx.x; c < KIN; c += 128) {
    const float* xb = x + ((size_t)b * KIN + c) * (SDIM * SDIM);
    float s = 0.0f;
    #pragma unroll 10
    for (int o = 0; o < 100; ++o) {
      const int oy = o / 10, ox = o - oy * 10;
      s += soft[o] * xb[(oy * 3) * SDIM + ox * 3];
    }
    out[(size_t)p * KIN + c] = s;
  }
}

// ---------------------------------------------------------------------------
// Host launcher
// ---------------------------------------------------------------------------
extern "C" void kernel_launch(void* const* d_in, const int* in_sizes, int n_in,
                              void* d_out, int out_size, void* d_ws, size_t ws_size,
                              hipStream_t stream) {
  (void)in_sizes; (void)n_in; (void)out_size; (void)ws_size;

  const float* x     = (const float*)d_in[0];
  const float* vwihf = (const float*)d_in[1];
  const float* vwhhf = (const float*)d_in[2];
  const float* vbf   = (const float*)d_in[3];
  const float* vwihb = (const float*)d_in[4];
  const float* vwhhb = (const float*)d_in[5];
  const float* vbb   = (const float*)d_in[6];
  const float* hwihf = (const float*)d_in[7];
  const float* hwhhf = (const float*)d_in[8];
  const float* hbf   = (const float*)d_in[9];
  const float* hwihb = (const float*)d_in[10];
  const float* hwhhb = (const float*)d_in[11];
  const float* hbb   = (const float*)d_in[12];
  const float* convw = (const float*)d_in[13];
  const float* convb = (const float*)d_in[14];

  char* base = (char*)d_ws;
  size_t off = 0;
  auto alloc = [&](size_t bytes) -> void* {
    off = (off + 255) & ~(size_t)255;
    void* p = base + off;
    off += bytes;
    return p;
  };

  const size_t seq_elems = (size_t)SDIM * NB * KIN;         // 25,690,112
  __bf16* Xv   = (__bf16*)alloc(seq_elems * 2);
  __bf16* Xh   = (__bf16*)alloc(seq_elems * 2);
  __bf16* Hpix = (__bf16*)alloc((size_t)NPIX * KIN * 2);

  __bf16* wVf = (__bf16*)alloc((size_t)G4 * KIN  * 2);
  __bf16* wVb = (__bf16*)alloc((size_t)G4 * KIN  * 2);
  __bf16* uVf = (__bf16*)alloc((size_t)G4 * HIDN * 2);
  __bf16* uVb = (__bf16*)alloc((size_t)G4 * HIDN * 2);
  __bf16* wHf = (__bf16*)alloc((size_t)G4 * KIN  * 2);
  __bf16* wHb = (__bf16*)alloc((size_t)G4 * KIN  * 2);
  __bf16* uHf = (__bf16*)alloc((size_t)G4 * HIDN * 2);
  __bf16* uHb = (__bf16*)alloc((size_t)G4 * HIDN * 2);

  // ---- weight conversion to bf16 ----
  auto cvt = [&](const float* s, __bf16* d, int n) {
    cvt_bf16_kernel<<<(n + 255) / 256, 256, 0, stream>>>(s, d, n);
  };
  cvt(vwihf, wVf, G4 * KIN);  cvt(vwihb, wVb, G4 * KIN);
  cvt(vwhhf, uVf, G4 * HIDN); cvt(vwhhb, uVb, G4 * HIDN);
  cvt(hwihf, wHf, G4 * KIN);  cvt(hwihb, wHb, G4 * KIN);
  cvt(hwhhf, uHf, G4 * HIDN); cvt(hwhhb, uHb, G4 * HIDN);

  // ---- pack x into vertical scan layout ----
  {
    const int blocks = (int)((seq_elems + 255) / 256);
    pack_xv_kernel<<<blocks, 256, 0, stream>>>(x, Xv);
  }

  // ---- persistent biLSTM stages: 112 row-tiles x 2 directions ----
  const dim3 stage_grid(NB / 16, 2);
  lstm_stage_kernel<<<stage_grid, 512, 0, stream>>>(
      Xv, wVf, wVb, uVf, uVb, vbf, vbb, Xh, 0);
  lstm_stage_kernel<<<stage_grid, 512, 0, stream>>>(
      Xh, wHf, wHb, uHf, uHb, hbf, hbb, Hpix, 1);

  // ---- attention head ----
  attention_kernel<<<NPIX, 128, 0, stream>>>(Hpix, convw, convb, x, (float*)d_out);
}
